// GroupedQueryAttention_8323646620084
// MI455X (gfx1250) — matrix operand order repaired and checked by reading, not compiled, not run on hardware
//
#include <hip/hip_runtime.h>
#include <hip/hip_bf16.h>
#include <stdint.h>

typedef __attribute__((ext_vector_type(16))) _Float16 v16h;
typedef __attribute__((ext_vector_type(8)))  float    v8f;
typedef __attribute__((ext_vector_type(4)))  unsigned v4u;
typedef __attribute__((ext_vector_type(8)))  int      v8i;
typedef __attribute__((ext_vector_type(4)))  int      v4i;

#define B_   4
#define T_   2048
#define C_   2048
#define H_   16
#define HKV_ 4
#define D_   128
#define NQKV (C_ + 2 * (HKV_ * D_))   /* 3072 */
#define SCALE_QK 0.08838834764831845f /* 1/sqrt(128) */

union FragH { v16h v; unsigned u[8]; };

// K-index base for the f16 A-matrix 16x32 VGPR layout (ISA 7.12.2):
// lanes 0-15 : v0..3 -> K=0..7,  v4..7 -> K=16..23
// lanes 16-31: v0..3 -> K=8..15, v4..7 -> K=24..31
__device__ __forceinline__ int kbaseA(int v, int h) {
    return h * 8 + (v < 4 ? 2 * v : 16 + 2 * (v - 4));
}

// ---------------------------------------------------------------------------
// Tensor Data Mover: 2-D tile DMA global -> LDS (ISA ch. 7/8, async_tensor).
// Moves tile_rows rows of tile_dw dwords each; source rows are
// row_stride_dw dwords apart; packed densely into LDS at lds_byte_off.
// D# group0: count=1, lds_addr, global_addr[56:0], type=2.
// D# group1: data_size=4B (code 2), tensor_dim0/1 = tile dims (never OOB),
//            tile_dim0/1, tensor_dim0_stride = row_stride_dw.
// ---------------------------------------------------------------------------
__device__ __forceinline__ void tdm_load_2d(
    unsigned lds_byte_off, const void* gp,
    unsigned tile_dw, unsigned tile_rows, unsigned row_stride_dw)
{
    unsigned long long ga = (unsigned long long)(uintptr_t)gp;
    v4u g0;
    g0[0] = 1u;                                   // count=1, user mode
    g0[1] = lds_byte_off;                         // lds_addr (bytes)
    g0[2] = (unsigned)(ga & 0xffffffffu);         // global_addr[31:0]
    g0[3] = (unsigned)((ga >> 32) & 0x1ffffffu)   // global_addr[56:32]
            | (2u << 30);                         // type = 2 ("image")
    unsigned td0 = tile_dw;                       // tensor_dim0 (dwords)
    unsigned td1 = tile_rows;                     // tensor_dim1
    unsigned long long s0 = row_stride_dw;        // tensor_dim0_stride
    v8i g1;
    g1[0] = (int)(2u << 16);                      // wg_mask=0, data_size=4B
    g1[1] = (int)((td0 & 0xffffu) << 16);         // tensor_dim0[15:0]
    g1[2] = (int)(((td0 >> 16) & 0xffffu) | ((td1 & 0xffffu) << 16));
    g1[3] = (int)(((td1 >> 16) & 0xffffu) | ((tile_dw & 0xffffu) << 16));
    g1[4] = (int)(tile_rows & 0xffffu);           // tile_dim1; tile_dim2=0
    g1[5] = (int)(s0 & 0xffffffffu);              // stride0[31:0]
    g1[6] = (int)((s0 >> 32) & 0xffffu);          // stride0[47:32], stride1=0
    g1[7] = 0;
    v4i z4 = {0, 0, 0, 0};
#if __has_include(<hip/amd_detail/amd_gfx1250_TDM.h>)
    v8i z8 = {0, 0, 0, 0, 0, 0, 0, 0};
    __builtin_amdgcn_tensor_load_to_lds(g0, g1, z4, z4, z8, 0);
#else
    __builtin_amdgcn_tensor_load_to_lds(g0, g1, z4, z4, 0);
#endif
}

__global__ void cvt_f32_to_f16(const float* __restrict__ s,
                               _Float16* __restrict__ d, int n) {
    int i = blockIdx.x * blockDim.x + threadIdx.x;
    if (i < n) d[i] = (_Float16)s[i];
}

// ---------------------------------------------------------------------------
// Tiled f16 WMMA GEMM:  Y(M,N) = A(M,K) * Bm(N,K)^T, fp32 accumulate.
// Block: 256 threads = 8 waves, tile 128x64, per-wave 32x32 (2x2 WMMA accs).
// Tile staging via TDM (tensor_load_to_lds), double-buffered in LDS.
// mode 0: write fp32 Y to outF.
// mode 1: QKV epilogue — RoPE on q/k column pairs, f16 outputs to qb/kb/vb.
// ---------------------------------------------------------------------------
__global__ __launch_bounds__(256) void gemm_f16_wmma(
    const _Float16* __restrict__ A, const _Float16* __restrict__ Bm,
    int M, int N, int K, int mode,
    float* __restrict__ outF,
    _Float16* __restrict__ qb, _Float16* __restrict__ kb,
    _Float16* __restrict__ vb,
    const float* __restrict__ cosT, const float* __restrict__ sinT)
{
    __shared__ _Float16 sA[2][128 * 32];
    __shared__ _Float16 sB[2][64 * 32];
    __shared__ float    sC[8 * 32 * 32];

    const int tid  = threadIdx.x;
    const int lane = tid & 31;
    const int wave = tid >> 5;
    const int nl   = lane & 15;
    const int hh   = lane >> 4;
    const int wm   = wave >> 1;          // 0..3 (M direction)
    const int wn   = wave & 1;           // 0..1 (N direction)
    const int n0   = blockIdx.x * 64;
    const int m0   = blockIdx.y * 128;

    v8f acc[2][2];
#pragma unroll
    for (int i = 0; i < 2; ++i)
#pragma unroll
        for (int j = 0; j < 2; ++j)
#pragma unroll
            for (int r = 0; r < 8; ++r) acc[i][j][r] = 0.0f;

    // prologue: DMA first K-tile into buffer 0
    if (tid == 0) {
        tdm_load_2d((unsigned)(uintptr_t)&sA[0][0],
                    A + (size_t)m0 * K, 16, 128, (unsigned)(K / 2));
        tdm_load_2d((unsigned)(uintptr_t)&sB[0][0],
                    Bm + (size_t)n0 * K, 16, 64, (unsigned)(K / 2));
        __builtin_amdgcn_s_wait_tensorcnt(0);
    }
    __syncthreads();

    int p = 0;
    for (int k0 = 0; k0 < K; k0 += 32) {
        const bool more = (k0 + 32) < K;
        if (more && tid == 0) {   // prefetch next K-tile into the other buffer
            tdm_load_2d((unsigned)(uintptr_t)&sA[1 - p][0],
                        A + (size_t)m0 * K + (k0 + 32), 16, 128,
                        (unsigned)(K / 2));
            tdm_load_2d((unsigned)(uintptr_t)&sB[1 - p][0],
                        Bm + (size_t)n0 * K + (k0 + 32), 16, 64,
                        (unsigned)(K / 2));
        }

        FragH af[2], bf[2];
#pragma unroll
        for (int s = 0; s < 2; ++s)
#pragma unroll
            for (int v = 0; v < 8; ++v)
                af[s].u[v] = *(const unsigned*)
                    &sA[p][(wm * 32 + s * 16 + nl) * 32 + kbaseA(v, hh)];
#pragma unroll
        for (int s = 0; s < 2; ++s)
#pragma unroll
            for (int v = 0; v < 8; ++v)
                bf[s].u[v] = *(const unsigned*)
                    &sB[p][(wn * 32 + s * 16 + nl) * 32 + kbaseA(v, hh)];

#pragma unroll
        for (int i = 0; i < 2; ++i)
#pragma unroll
            for (int j = 0; j < 2; ++j)
                acc[i][j] = __builtin_amdgcn_wmma_f32_16x16x32_f16(
                    false, af[i].v, false, bf[j].v,
                    (short)0, acc[i][j], false, false);

        if (more && tid == 0) __builtin_amdgcn_s_wait_tensorcnt(0);
        __syncthreads();
        p ^= 1;
    }

    if (mode == 0) {
#pragma unroll
        for (int i = 0; i < 2; ++i)
#pragma unroll
            for (int j = 0; j < 2; ++j)
#pragma unroll
                for (int r = 0; r < 8; ++r) {
                    int row = m0 + wm * 32 + i * 16 + hh * 8 + r;
                    int col = n0 + wn * 32 + j * 16 + nl;
                    outF[(size_t)row * N + col] = acc[i][j][r];
                }
    } else {
        // stage the 32x32 fp32 tile per wave, then apply RoPE on column pairs
        float* cw = &sC[wave * 1024];
#pragma unroll
        for (int i = 0; i < 2; ++i)
#pragma unroll
            for (int j = 0; j < 2; ++j)
#pragma unroll
                for (int r = 0; r < 8; ++r)
                    cw[(i * 16 + hh * 8 + r) * 32 + j * 16 + nl] = acc[i][j][r];
        __syncthreads();

        for (int p2 = lane; p2 < 512; p2 += 32) {
            int row = p2 >> 4;
            int pr  = p2 & 15;
            int c0  = pr * 2;
            float e = cw[row * 32 + c0];
            float o = cw[row * 32 + c0 + 1];
            int gm = m0 + wm * 32 + row;
            int b  = gm >> 11;               // T_ = 2048
            int t  = gm & (T_ - 1);
            int n  = n0 + wn * 32 + c0;
            int d  = n & (D_ - 1);
            int dh = d >> 1;
            if (n < C_) {                    // Q + RoPE
                int hq = n >> 7;
                float c = cosT[t * (D_ / 2) + dh];
                float s = sinT[t * (D_ / 2) + dh];
                size_t base = ((size_t)(b * H_ + hq) * T_ + t) * D_ + d;
                qb[base]     = (_Float16)(e * c - o * s);
                qb[base + 1] = (_Float16)(e * s + o * c);
            } else if (n < C_ + HKV_ * D_) { // K + RoPE
                int kh = (n - C_) >> 7;
                float c = cosT[t * (D_ / 2) + dh];
                float s = sinT[t * (D_ / 2) + dh];
                size_t base = ((size_t)(b * HKV_ + kh) * T_ + t) * D_ + d;
                kb[base]     = (_Float16)(e * c - o * s);
                kb[base + 1] = (_Float16)(e * s + o * c);
            } else {                         // V passthrough
                int vh = (n - C_ - HKV_ * D_) >> 7;
                size_t base = ((size_t)(b * HKV_ + vh) * T_ + t) * D_ + d;
                vb[base]     = (_Float16)e;
                vb[base + 1] = (_Float16)o;
            }
        }
    }
}

// ---------------------------------------------------------------------------
// Flash attention with WMMA. Block: 128 threads = 4 waves; block handles 64
// query rows of one (b, h); each wave owns a 16-row q tile; key tiles of 32.
// Q and K tiles staged via TDM; V staged cooperatively (transposed in LDS).
// ---------------------------------------------------------------------------
__global__ __launch_bounds__(128) void gqa_flash_wmma(
    const _Float16* __restrict__ Qb, const _Float16* __restrict__ Kb,
    const _Float16* __restrict__ Vb, _Float16* __restrict__ attnA)
{
    __shared__ _Float16 sQ[64 * 128];   // [qrow][d]
    __shared__ _Float16 sK[32 * 128];   // [key][d]
    __shared__ _Float16 sVt[128 * 32];  // [d][key]  (transposed for B layout)
    __shared__ _Float16 sP[4 * 16 * 32];// per-wave P tile [m][k]

    const int tid  = threadIdx.x;
    const int lane = tid & 31;
    const int wave = tid >> 5;
    const int nl   = lane & 15;
    const int hh   = lane >> 4;
    const int bh   = blockIdx.x;
    const int b    = bh >> 4;
    const int hq   = bh & 15;
    const int kvh  = hq >> 2;            // G = 4
    const int qt0  = blockIdx.y * 64;

    const _Float16* Qp = Qb + (size_t)(b * H_ + hq) * T_ * D_;
    const _Float16* Kp = Kb + (size_t)(b * HKV_ + kvh) * T_ * D_;
    const _Float16* Vp = Vb + (size_t)(b * HKV_ + kvh) * T_ * D_;

    // Q tile 64x128 via TDM (rows contiguous, 64 dwords each)
    if (tid == 0) {
        tdm_load_2d((unsigned)(uintptr_t)&sQ[0],
                    Qp + (size_t)qt0 * D_, 64, 64, 64);
        __builtin_amdgcn_s_wait_tensorcnt(0);
    }
    __syncthreads();

    FragH qf[4];
#pragma unroll
    for (int kc = 0; kc < 4; ++kc)
#pragma unroll
        for (int v = 0; v < 8; ++v)
            qf[kc].u[v] = *(const unsigned*)
                &sQ[(wave * 16 + nl) * 128 + kc * 32 + kbaseA(v, hh)];

    float mrow[8], lrow[8];
    v8f accO[8];
#pragma unroll
    for (int r = 0; r < 8; ++r) { mrow[r] = -3.0e30f; lrow[r] = 0.0f; }
#pragma unroll
    for (int nc = 0; nc < 8; ++nc)
#pragma unroll
        for (int r = 0; r < 8; ++r) accO[nc][r] = 0.0f;

    const int kend = qt0 + 64;               // causal bound for this block
    const int qrb  = qt0 + wave * 16 + hh * 8;

    for (int kt = 0; kt < kend; kt += 32) {
        __syncthreads();
        // K tile 32x128 via TDM
        if (tid == 0)
            tdm_load_2d((unsigned)(uintptr_t)&sK[0],
                        Kp + (size_t)kt * D_, 64, 32, 64);
        // V tile 32x128, stored transposed [d][key]
#pragma unroll
        for (int it = 0; it < 16; ++it) {
            int i = tid + it * 128;
            int row = i >> 6, c2 = (i & 63) * 2;
            union { unsigned u; _Float16 hv[2]; } cv;
            cv.u = *(const unsigned*)&Vp[(size_t)(kt + row) * D_ + c2];
            sVt[c2 * 32 + row]       = cv.hv[0];
            sVt[(c2 + 1) * 32 + row] = cv.hv[1];
        }
        if (tid == 0) __builtin_amdgcn_s_wait_tensorcnt(0);
        __syncthreads();

        // S = Q * K^T : two 16x16 C tiles (keys 0-15 and 16-31 of this kt)
        v8f s0, s1;
#pragma unroll
        for (int r = 0; r < 8; ++r) { s0[r] = 0.0f; s1[r] = 0.0f; }
#pragma unroll
        for (int kc = 0; kc < 4; ++kc) {
            FragH kf0, kf1;
#pragma unroll
            for (int v = 0; v < 8; ++v) {
                kf0.u[v] = *(const unsigned*)
                    &sK[nl * 128 + kc * 32 + kbaseA(v, hh)];
                kf1.u[v] = *(const unsigned*)
                    &sK[(16 + nl) * 128 + kc * 32 + kbaseA(v, hh)];
            }
            s0 = __builtin_amdgcn_wmma_f32_16x16x32_f16(
                false, qf[kc].v, false, kf0.v, (short)0, s0, false, false);
            s1 = __builtin_amdgcn_wmma_f32_16x16x32_f16(
                false, qf[kc].v, false, kf1.v, (short)0, s1, false, false);
        }

        // online softmax over 32 keys
#pragma unroll
        for (int r = 0; r < 8; ++r) {
            int qr = qrb + r;
            float a0 = s0[r] * SCALE_QK;
            float a1 = s1[r] * SCALE_QK;
            bool ok0 = (kt + nl) <= qr;
            bool ok1 = (kt + 16 + nl) <= qr;
            if (!ok0) a0 = -3.0e30f;
            if (!ok1) a1 = -3.0e30f;
            float mx = fmaxf(a0, a1);
            mx = fmaxf(mx, __shfl_xor(mx, 1, 16));
            mx = fmaxf(mx, __shfl_xor(mx, 2, 16));
            mx = fmaxf(mx, __shfl_xor(mx, 4, 16));
            mx = fmaxf(mx, __shfl_xor(mx, 8, 16));
            float mnew = fmaxf(mrow[r], mx);
            float so   = __expf(mrow[r] - mnew);
            float p0 = ok0 ? __expf(a0 - mnew) : 0.0f;
            float p1 = ok1 ? __expf(a1 - mnew) : 0.0f;
            float rs = p0 + p1;
            rs += __shfl_xor(rs, 1, 16);
            rs += __shfl_xor(rs, 2, 16);
            rs += __shfl_xor(rs, 4, 16);
            rs += __shfl_xor(rs, 8, 16);
            mrow[r] = mnew;
            lrow[r] = lrow[r] * so + rs;
            sP[wave * 512 + (hh * 8 + r) * 32 + nl]      = (_Float16)p0;
            sP[wave * 512 + (hh * 8 + r) * 32 + 16 + nl] = (_Float16)p1;
#pragma unroll
            for (int nc = 0; nc < 8; ++nc) accO[nc][r] *= so;
        }

        // wave-local LDS fence before re-reading sP in A layout
        asm volatile("s_wait_dscnt 0" ::: "memory");

        FragH pf;
#pragma unroll
        for (int v = 0; v < 8; ++v)
            pf.u[v] = *(const unsigned*)
                &sP[wave * 512 + nl * 32 + kbaseA(v, hh)];

        // O += P * V
#pragma unroll
        for (int nc = 0; nc < 8; ++nc) {
            FragH vf;
#pragma unroll
            for (int v = 0; v < 8; ++v)
                vf.u[v] = *(const unsigned*)
                    &sVt[(nc * 16 + nl) * 32 + kbaseA(v, hh)];
            accO[nc] = __builtin_amdgcn_wmma_f32_16x16x32_f16(
                false, pf.v, false, vf.v, (short)0, accO[nc], false, false);
        }
    }

    // normalize + store f16 attention output [b][t][h*D + d]
#pragma unroll
    for (int r = 0; r < 8; ++r) {
        float inv = lrow[r] > 0.0f ? 1.0f / lrow[r] : 0.0f;
        int t = qt0 + wave * 16 + hh * 8 + r;
#pragma unroll
        for (int nc = 0; nc < 8; ++nc) {
            int col = hq * D_ + nc * 16 + nl;
            attnA[(size_t)(b * T_ + t) * C_ + col] =
                (_Float16)(accO[nc][r] * inv);
        }
    }
}

extern "C" void kernel_launch(void* const* d_in, const int* in_sizes, int n_in,
                              void* d_out, int out_size, void* d_ws,
                              size_t ws_size, hipStream_t stream)
{
    const float* x    = (const float*)d_in[0];
    const float* Wq   = (const float*)d_in[1];
    const float* Wk   = (const float*)d_in[2];
    const float* Wv   = (const float*)d_in[3];
    const float* Wo   = (const float*)d_in[4];
    const float* cosT = (const float*)d_in[5];
    const float* sinT = (const float*)d_in[6];
    float* out = (float*)d_out;

    char* w = (char*)d_ws;
    auto carve = [&](size_t bytes) {
        char* p = w;
        w += (bytes + 255) & ~(size_t)255;
        return p;
    };
    _Float16* Xh    = (_Float16*)carve((size_t)B_ * T_ * C_ * 2);
    _Float16* Wqkvh = (_Float16*)carve((size_t)NQKV * C_ * 2);
    _Float16* Woh   = (_Float16*)carve((size_t)C_ * C_ * 2);
    _Float16* Qb    = (_Float16*)carve((size_t)B_ * H_ * T_ * D_ * 2);
    _Float16* Kb    = (_Float16*)carve((size_t)B_ * HKV_ * T_ * D_ * 2);
    _Float16* Vb    = (_Float16*)carve((size_t)B_ * HKV_ * T_ * D_ * 2);
    _Float16* Ah    = (_Float16*)carve((size_t)B_ * T_ * C_ * 2);

    auto cvt = [&](const float* s, _Float16* d, int n) {
        cvt_f32_to_f16<<<(n + 255) / 256, 256, 0, stream>>>(s, d, n);
    };
    cvt(x, Xh, B_ * T_ * C_);
    cvt(Wq, Wqkvh, C_ * C_);
    cvt(Wk, Wqkvh + (size_t)C_ * C_, HKV_ * D_ * C_);
    cvt(Wv, Wqkvh + (size_t)(C_ + HKV_ * D_) * C_, HKV_ * D_ * C_);
    cvt(Wo, Woh, C_ * C_);

    // QKV projection + RoPE epilogue
    dim3 g1(NQKV / 64, (B_ * T_) / 128);
    gemm_f16_wmma<<<g1, 256, 0, stream>>>(Xh, Wqkvh, B_ * T_, NQKV, C_, 1,
                                          nullptr, Qb, Kb, Vb, cosT, sinT);

    // flash attention
    dim3 g2(B_ * H_, T_ / 64);
    gqa_flash_wmma<<<g2, 128, 0, stream>>>(Qb, Kb, Vb, Ah);

    // output projection -> fp32 d_out
    dim3 g3(C_ / 64, (B_ * T_) / 128);
    gemm_f16_wmma<<<g3, 256, 0, stream>>>(Ah, Woh, B_ * T_, C_, C_, 0,
                                          out, nullptr, nullptr, nullptr,
                                          nullptr, nullptr);

    (void)in_sizes; (void)n_in; (void)out_size; (void)ws_size;
}
